// LiquidPhys_87617332838407
// MI455X (gfx1250) — matrix operand order repaired and checked by reading, not compiled, not run on hardware
//
#include <hip/hip_runtime.h>

typedef __attribute__((ext_vector_type(16))) _Float16 v16h;
typedef __attribute__((ext_vector_type(8)))  _Float16 v8h;
typedef __attribute__((ext_vector_type(8)))  float    v8f;

__device__ __forceinline__ float siluf(float x) { return x / (1.f + __expf(-x)); }
__device__ __forceinline__ float sigmf(float x) { return 1.f / (1.f + __expf(-x)); }

__device__ __forceinline__ v8h zero8h() {
  v8h r;
#pragma unroll
  for (int i = 0; i < 8; ++i) r[i] = (_Float16)0.f;
  return r;
}

// ---- CDNA5 async global->LDS staging (ASYNCcnt path, inline asm) ----------
// global_load_async_to_lds_b128: VDST = per-lane LDS byte offset, VADDR = 64-bit
// global address. Tracked by ASYNCcnt; drained with s_wait_asynccnt 0.
#define USE_ASYNC_ASM 1

__device__ __forceinline__ void lds_cp16(_Float16* dst, const _Float16* src) {
#if USE_ASYNC_ASM
  unsigned lds_off = (unsigned)(unsigned long long)(void*)dst;
  unsigned long long gaddr = (unsigned long long)(const void*)src;
  asm volatile("global_load_async_to_lds_b128 %0, %1, off"
               :
               : "v"(lds_off), "v"(gaddr)
               : "memory");
#else
  *(v8h*)dst = *(const v8h*)src;
#endif
}

__device__ __forceinline__ void lds_cp_wait() {
#if USE_ASYNC_ASM
#if __has_builtin(__builtin_amdgcn_s_wait_asynccnt)
  __builtin_amdgcn_s_wait_asynccnt(0);
#else
  asm volatile("s_wait_asynccnt 0x0" ::: "memory");
#endif
#endif
}

// ---------------------------------------------------------------------------
// Prep: frames [4,6,160,72,72] f32 -> padded channel-last f16 [4,162,76,76,8]
// ---------------------------------------------------------------------------
__global__ void pack_frames(const float* __restrict__ fr, _Float16* __restrict__ out) {
  long idx = (long)blockIdx.x * blockDim.x + threadIdx.x;
  const long total = 4L * 162 * 76 * 76;
  if (idx >= total) return;
  int w = (int)(idx % 76);
  int h = (int)((idx / 76) % 76);
  int t = (int)((idx / (76 * 76)) % 162);
  int b = (int)(idx / (76L * 76 * 162));
  v8h v = zero8h();
  if (t >= 1 && t <= 160 && h >= 2 && h <= 73 && w >= 2 && w <= 73) {
    int t0 = t - 1, h0 = h - 2, w0 = w - 2;
#pragma unroll
    for (int c = 0; c < 6; ++c)
      v[c] = (_Float16)fr[(((size_t)b * 6 + c) * 160 + t0) * 72 * 72 + (size_t)h0 * 72 + w0];
  }
  *(v8h*)(out + idx * 8) = v;
}

__global__ void zero16(uint4* p, long n16) {
  long i = (long)blockIdx.x * blockDim.x + threadIdx.x;
  if (i < n16) { uint4 z; z.x = z.y = z.z = z.w = 0u; p[i] = z; }
}

// ---------------------------------------------------------------------------
// Pack conv weights (BN folded) into WMMA A-fragments, contiguous k order (k = 16*(lane/16) + e).
// ---------------------------------------------------------------------------
__global__ void pack_w(const float* __restrict__ w, const float* __restrict__ g,
                       const float* __restrict__ vv, _Float16* __restrict__ wpack,
                       int O, int CI, int CIPAD, int KT, int KH, int KW, int NCHUNKS) {
  long idx = (long)blockIdx.x * blockDim.x + threadIdx.x;
  long total = (long)(O / 16) * NCHUNKS * 512;
  if (idx >= total) return;
  int e = (int)(idx & 15);
  int lane = (int)((idx >> 4) & 31);
  int c = (int)((idx >> 9) % NCHUNKS);
  int mt = (int)(idx / (512L * NCHUNKS));
  int m = lane & 15;
  int kl = e + ((lane >> 4) << 4);
  int K = c * 32 + kl;
  int tap = K / CIPAD;
  int ci = K % CIPAD;
  int o = mt * 16 + m;
  float val = 0.f;
  if (tap < KT * KH * KW && ci < CI) {
    int kt = tap / (KH * KW);
    int r = tap % (KH * KW);
    int kh = r / KW, kw = r % KW;
    float s = g[o] * rsqrtf(vv[o] + 1e-5f);
    val = w[(((size_t)o * CI + ci) * KT + kt) * KH * KW + (size_t)kh * KW + kw] * s;
  }
  wpack[idx] = (_Float16)val;
}

__global__ void bias_k(const float* __restrict__ g, const float* __restrict__ bb,
                       const float* __restrict__ m, const float* __restrict__ v,
                       float* __restrict__ o, int n) {
  int i = blockIdx.x * blockDim.x + threadIdx.x;
  if (i < n) { float s = g[i] * rsqrtf(v[i] + 1e-5f); o[i] = bb[i] - m[i] * s; }
}

#define WMMA16(A, B, C) \
  __builtin_amdgcn_wmma_f32_16x16x32_f16(false, (A), false, (B), (short)0, (C), false, false)

// ---------------------------------------------------------------------------
// conv1: 6->32ch, k(3,5,5) pad(1,2,2) + BN + SiLU + 2x2 avgpool.
// Block = 64 thr (2 waves), computes full W row pair (h, h+1) for one (b,t).
// Patch LDS: [kt 3][hh 6][w 84(pad)][ci 8]; valid region 3x6x76x8.
// ---------------------------------------------------------------------------
__global__ void __launch_bounds__(64) conv1_k(const _Float16* __restrict__ in,
                                              const _Float16* __restrict__ wp,
                                              const float* __restrict__ bias,
                                              _Float16* __restrict__ out) {
  __shared__ __align__(64) _Float16 patch[3 * 6 * 84 * 8];
  __shared__ __align__(32) _Float16 psh[2][2][80][16];
  int gid = blockIdx.x;
  int ht = gid % 36;
  int t = (gid / 36) % 160;
  int b = gid / (36 * 160);
  int hBase = ht * 2;
  int tid = threadIdx.x, lane = tid & 31, mt = tid >> 5, hi = lane >> 4, n = lane & 15;

  for (int g = tid; g < 1368; g += 64) {  // 3*6*76 v8h voxels
    int w = g % 76;
    int hh = (g / 76) % 6;
    int kt = g / 456;
    size_t src = (((size_t)(b * 162 + t + kt)) * 76 + (hBase + hh)) * 76 + w;
    int dst = ((kt * 6 + hh) * 84 + w) * 8;
    lds_cp16(&patch[dst], in + src * 8);
  }
  lds_cp_wait();
  __syncthreads();

  v8f acc[2][5] = {};
#pragma unroll
  for (int c = 0; c < 19; ++c) {
    v16h afrag = *(const v16h*)(wp + ((size_t)(mt * 19 + c) * 32 + lane) * 16);
    int t0 = c * 4 + hi * 2;        // tap for fragment elements e<8
    int t1 = t0 + 1;                // tap for e>=8
    bool inv1 = (t1 > 74);          // tap 75 is K-padding
    int t1c = inv1 ? 74 : t1;
    int kt0 = t0 / 25, r0 = t0 % 25, kh0 = r0 / 5, kw0 = r0 % 5;
    int kt1 = t1c / 25, r1 = t1c % 25, kh1 = r1 / 5, kw1 = r1 % 5;
#pragma unroll
    for (int r = 0; r < 2; ++r) {
#pragma unroll
      for (int nt = 0; nt < 5; ++nt) {
        int wn = nt * 16 + n;
        v8h lo = *(const v8h*)&patch[((kt0 * 6 + r + kh0) * 84 + (wn + kw0)) * 8];
        v8h hs = *(const v8h*)&patch[((kt1 * 6 + r + kh1) * 84 + (wn + kw1)) * 8];
        if (inv1) hs = zero8h();
        v16h bf;
        ((v8h*)&bf)[0] = lo;
        ((v8h*)&bf)[1] = hs;
        acc[r][nt] = WMMA16(afrag, bf, acc[r][nt]);
      }
    }
  }

  // bias + SiLU into LDS, then 2x2 avg-pool -> act1 padded [4,162,38,38,32]
#pragma unroll
  for (int r = 0; r < 2; ++r)
#pragma unroll
    for (int nt = 0; nt < 5; ++nt)
#pragma unroll
      for (int j = 0; j < 8; ++j) {
        int cl = hi * 8 + j;
        float v = acc[r][nt][j] + bias[mt * 16 + cl];
        psh[mt][r][nt * 16 + n][cl] = (_Float16)siluf(v);
      }
  __syncthreads();
  for (int q = tid; q < 36 * 4; q += 64) {
    int np = q >> 2;  // pooled w 0..35
    int gq = q & 3;
    int mt2 = gq >> 1, gg = gq & 1;
    v8h ov;
#pragma unroll
    for (int cc = 0; cc < 8; ++cc) {
      int cl = gg * 8 + cc;
      float s = (float)psh[mt2][0][2 * np][cl] + (float)psh[mt2][0][2 * np + 1][cl] +
                (float)psh[mt2][1][2 * np][cl] + (float)psh[mt2][1][2 * np + 1][cl];
      ov[cc] = (_Float16)(0.25f * s);
    }
    size_t oidx = ((((size_t)b * 162 + (t + 1)) * 38 + (ht + 1)) * 38 + (np + 1)) * 32 + mt2 * 16 + gg * 8;
    *(v8h*)(out + oidx) = ov;
  }
}

// ---------------------------------------------------------------------------
// conv2: 32->64ch, k(3,3,3) pad 1 + BN + SiLU + 2x2 avgpool.
// Block = 128 thr (4 waves), full 36-wide row pair. Patch [3][4][50(pad)][32].
// ---------------------------------------------------------------------------
__global__ void __launch_bounds__(128) conv2_k(const _Float16* __restrict__ in,
                                               const _Float16* __restrict__ wp,
                                               const float* __restrict__ bias,
                                               _Float16* __restrict__ out) {
  __shared__ __align__(64) _Float16 patch[3 * 4 * 50 * 32];
  __shared__ __align__(32) _Float16 psh[4][2][48][16];
  int gid = blockIdx.x;
  int ht = gid % 18;
  int t = (gid / 18) % 160;
  int b = gid / (18 * 160);
  int hBase = ht * 2;
  int tid = threadIdx.x, lane = tid & 31, mt = tid >> 5, hi = lane >> 4, n = lane & 15;

  for (int g = tid; g < 1824; g += 128) {  // 3*4*38*4 v8h
    int ci8 = g & 3;
    int w = (g >> 2) % 38;
    int hh = (g / 152) % 4;
    int kt = g / 608;
    size_t src = ((((size_t)(b * 162 + t + kt)) * 38 + (hBase + hh)) * 38 + w) * 32 + ci8 * 8;
    int dst = ((kt * 4 + hh) * 50 + w) * 32 + ci8 * 8;
    lds_cp16(&patch[dst], in + src);
  }
  lds_cp_wait();
  __syncthreads();

  v8f acc[2][3] = {};
#pragma unroll
  for (int c = 0; c < 27; ++c) {
    int kt = c / 9, kh = (c / 3) % 3, kw = c % 3;  // compile-time per unrolled c
    v16h afrag = *(const v16h*)(wp + ((size_t)(mt * 27 + c) * 32 + lane) * 16);
#pragma unroll
    for (int r = 0; r < 2; ++r)
#pragma unroll
      for (int nt = 0; nt < 3; ++nt) {
        v16h bf = *(const v16h*)&patch[((kt * 4 + r + kh) * 50 + (nt * 16 + n + kw)) * 32 + hi * 16];
        acc[r][nt] = WMMA16(afrag, bf, acc[r][nt]);
      }
  }

#pragma unroll
  for (int r = 0; r < 2; ++r)
#pragma unroll
    for (int nt = 0; nt < 3; ++nt)
#pragma unroll
      for (int j = 0; j < 8; ++j) {
        int cl = hi * 8 + j;
        float v = acc[r][nt][j] + bias[mt * 16 + cl];
        psh[mt][r][nt * 16 + n][cl] = (_Float16)siluf(v);
      }
  __syncthreads();
  for (int q = tid; q < 18 * 8; q += 128) {
    int np = q >> 3;  // pooled w 0..17
    int gq = q & 7;
    int mt2 = gq >> 1, gg = gq & 1;
    v8h ov;
#pragma unroll
    for (int cc = 0; cc < 8; ++cc) {
      int cl = gg * 8 + cc;
      float s = (float)psh[mt2][0][2 * np][cl] + (float)psh[mt2][0][2 * np + 1][cl] +
                (float)psh[mt2][1][2 * np][cl] + (float)psh[mt2][1][2 * np + 1][cl];
      ov[cc] = (_Float16)(0.25f * s);
    }
    size_t oidx = ((((size_t)b * 162 + (t + 1)) * 20 + (ht + 1)) * 20 + (np + 1)) * 64 + mt2 * 16 + gg * 8;
    *(v8h*)(out + oidx) = ov;
  }
}

// ---------------------------------------------------------------------------
// conv3: 64->64ch, k(3,3,3) pad 1 + BN + SiLU (no pool).
// Block = 128 thr, one full 18-wide row. Patch [3][3][34(pad)][64].
// ---------------------------------------------------------------------------
__global__ void __launch_bounds__(128) conv3_k(const _Float16* __restrict__ in,
                                               const _Float16* __restrict__ wp,
                                               const float* __restrict__ bias,
                                               _Float16* __restrict__ out) {
  __shared__ __align__(64) _Float16 patch[3 * 3 * 34 * 64];
  int gid = blockIdx.x;
  int ht = gid % 18;
  int t = (gid / 18) % 160;
  int b = gid / (18 * 160);
  int tid = threadIdx.x, lane = tid & 31, mt = tid >> 5, hi = lane >> 4, n = lane & 15;

  for (int g = tid; g < 1440; g += 128) {  // 3*3*20*8 v8h
    int ci8 = g & 7;
    int w = (g >> 3) % 20;
    int hh = (g / 160) % 3;
    int kt = g / 480;
    size_t src = ((((size_t)(b * 162 + t + kt)) * 20 + (ht + hh)) * 20 + w) * 64 + ci8 * 8;
    int dst = ((kt * 3 + hh) * 34 + w) * 64 + ci8 * 8;
    lds_cp16(&patch[dst], in + src);
  }
  lds_cp_wait();
  __syncthreads();

  v8f acc[2] = {};
#pragma unroll
  for (int c = 0; c < 54; ++c) {
    int tap = c >> 1, cb = (c & 1) * 32;
    int kt = tap / 9, kh = (tap / 3) % 3, kw = tap % 3;
    v16h afrag = *(const v16h*)(wp + ((size_t)(mt * 54 + c) * 32 + lane) * 16);
#pragma unroll
    for (int nt = 0; nt < 2; ++nt) {
      v16h bf = *(const v16h*)&patch[((kt * 3 + kh) * 34 + (nt * 16 + n + kw)) * 64 + cb + hi * 16];
      acc[nt] = WMMA16(afrag, bf, acc[nt]);
    }
  }

#pragma unroll
  for (int nt = 0; nt < 2; ++nt) {
    int wv = nt * 16 + n;
    if (wv < 18) {
      v8h ov;
#pragma unroll
      for (int j = 0; j < 8; ++j) {
        float v = acc[nt][j] + bias[mt * 16 + hi * 8 + j];
        ov[j] = (_Float16)siluf(v);
      }
      size_t oidx = (((size_t)(b * 160 + t) * 18 + ht) * 18 + wv) * 64 + mt * 16 + hi * 8;
      *(v8h*)(out + oidx) = ov;
    }
  }
}

// ---------------------------------------------------------------------------
__global__ void spatial_mean(const _Float16* __restrict__ a3, float* __restrict__ enc) {
  int bt = blockIdx.x;
  int c = threadIdx.x;  // 64
  const _Float16* p = a3 + (size_t)bt * 324 * 64 + c;
  float s = 0.f;
  for (int i = 0; i < 324; ++i) s += (float)p[(size_t)i * 64];
  enc[(size_t)bt * 64 + c] = s * (1.f / 324.f);
}

__global__ void motion_k(const float* __restrict__ pose, const float* __restrict__ flow,
                         const float* __restrict__ lng, const float* __restrict__ lnb,
                         const float* __restrict__ w1, const float* __restrict__ b1,
                         const float* __restrict__ w2, const float* __restrict__ b2,
                         const float* __restrict__ gw, const float* __restrict__ gb,
                         float* __restrict__ me, float* __restrict__ gates) {
  int bt = blockIdx.x;
  int tid = threadIdx.x;
  __shared__ float mo[10], mn[10], h1[32];
  if (tid < 3) mo[tid] = pose[bt * 3 + tid];
  else if (tid < 10) mo[tid] = flow[bt * 7 + (tid - 3)];
  __syncthreads();
  if (tid == 0) {
    float mu = 0.f;
    for (int i = 0; i < 10; ++i) mu += mo[i];
    mu *= 0.1f;
    float var = 0.f;
    for (int i = 0; i < 10; ++i) { float d = mo[i] - mu; var += d * d; }
    var *= 0.1f;
    float rs = rsqrtf(var + 1e-5f);
    for (int i = 0; i < 10; ++i) mn[i] = (mo[i] - mu) * rs * lng[i] + lnb[i];
  }
  __syncthreads();
  if (tid < 32) {
    float s = b1[tid];
    for (int k = 0; k < 10; ++k) s += mn[k] * w1[k * 32 + tid];
    h1[tid] = siluf(s);
  }
  __syncthreads();
  if (tid < 16) {
    float s = b2[tid];
    for (int k = 0; k < 32; ++k) s += h1[k] * w2[k * 16 + tid];
    me[bt * 16 + tid] = siluf(s);
  }
  {
    float s = gb[tid];
    for (int k = 0; k < 10; ++k) s += mn[k] * gw[k * 64 + tid];
    gates[bt * 64 + tid] = sigmf(s);
  }
}

__global__ void cfc_k(const float* __restrict__ enc, const float* __restrict__ me,
                      const float* __restrict__ bbw, const float* __restrict__ bbb,
                      const float* __restrict__ f1w, const float* __restrict__ f1b,
                      const float* __restrict__ f2w, const float* __restrict__ f2b,
                      const float* __restrict__ taw, const float* __restrict__ tab,
                      const float* __restrict__ tbw, const float* __restrict__ tbb,
                      float* __restrict__ seq) {
  __shared__ float xin[4][144];
  __shared__ float z[4][128];
  int tid = threadIdx.x;
  { int b = tid >> 6, j = tid & 63; xin[b][80 + j] = 0.f; }
  for (int t = 0; t < 160; ++t) {
    for (int q = tid; q < 320; q += 256) {
      int b = q / 80, i = q % 80;
      xin[b][i] = (i < 64) ? enc[((size_t)b * 160 + t) * 64 + i]
                           : me[((size_t)b * 160 + t) * 16 + (i - 64)];
    }
    __syncthreads();
    for (int q = tid; q < 512; q += 256) {
      int b = q >> 7, j = q & 127;
      float s = bbb[j];
      for (int k = 0; k < 144; ++k) s += xin[b][k] * bbw[k * 128 + j];
      z[b][j] = 1.7159f * tanhf(0.666f * s);
    }
    __syncthreads();
    {
      int b = tid >> 6, j = tid & 63;
      float s1 = f1b[j], s2 = f2b[j], s3 = tab[j] + tbb[j];
      for (int k = 0; k < 128; ++k) {
        float zz = z[b][k];
        s1 += zz * f1w[k * 64 + j];
        s2 += zz * f2w[k * 64 + j];
        s3 += zz * (taw[k * 64 + j] + tbw[k * 64 + j]);
      }
      float ff1 = tanhf(s1), ff2 = tanhf(s2), ti = sigmf(s3);
      float hn = ff1 * (1.f - ti) + ti * ff2;
      seq[((size_t)t * 4 + b) * 64 + j] = hn;
      xin[b][80 + j] = hn;
    }
    __syncthreads();
  }
}

__global__ void gate_k(const float* __restrict__ seq, const float* __restrict__ gates,
                       float* __restrict__ gated) {
  int b = threadIdx.x >> 6, j = threadIdx.x & 63;
  float s = seq[(size_t)b * 64 + j];
  gated[((size_t)b * 160) * 64 + j] = s;
  for (int t = 1; t < 160; ++t) {
    float g = gates[((size_t)b * 160 + t) * 64 + j];
    float x = seq[((size_t)t * 4 + b) * 64 + j];
    s = g * x + (1.f - g) * s;
    gated[((size_t)b * 160 + t) * 64 + j] = s;
  }
}

__global__ void head_k(const float* __restrict__ gated, const float* __restrict__ w1,
                       const float* __restrict__ b1, const float* __restrict__ w2,
                       const float* __restrict__ b2, float* __restrict__ out) {
  int bt = blockIdx.x;
  int j = threadIdx.x;
  __shared__ float h[32];
  float s = b1[j];
  for (int k = 0; k < 64; ++k) s += gated[(size_t)bt * 64 + k] * w1[k * 32 + j];
  h[j] = siluf(s) * w2[j];
  __syncthreads();
  if (j == 0) {
    float r = b2[0];
    for (int k = 0; k < 32; ++k) r += h[k];
    out[bt] = r;
  }
}

// ---------------------------------------------------------------------------
extern "C" void kernel_launch(void* const* d_in, const int* in_sizes, int n_in,
                              void* d_out, int out_size, void* d_ws, size_t ws_size,
                              hipStream_t stream) {
  (void)in_sizes; (void)n_in; (void)out_size; (void)ws_size;
  const float* frames = (const float*)d_in[0];
  const float* pose = (const float*)d_in[1];
  const float* flow = (const float*)d_in[2];
  const float* conv1_w = (const float*)d_in[3];
  const float* bn1_g = (const float*)d_in[4];
  const float* bn1_b = (const float*)d_in[5];
  const float* bn1_m = (const float*)d_in[6];
  const float* bn1_v = (const float*)d_in[7];
  const float* conv2_w = (const float*)d_in[8];
  const float* bn2_g = (const float*)d_in[9];
  const float* bn2_b = (const float*)d_in[10];
  const float* bn2_m = (const float*)d_in[11];
  const float* bn2_v = (const float*)d_in[12];
  const float* conv3_w = (const float*)d_in[13];
  const float* bn3_g = (const float*)d_in[14];
  const float* bn3_b = (const float*)d_in[15];
  const float* bn3_m = (const float*)d_in[16];
  const float* bn3_v = (const float*)d_in[17];
  const float* ln_g = (const float*)d_in[18];
  const float* ln_b = (const float*)d_in[19];
  const float* mlp_w1 = (const float*)d_in[20];
  const float* mlp_b1 = (const float*)d_in[21];
  const float* mlp_w2 = (const float*)d_in[22];
  const float* mlp_b2 = (const float*)d_in[23];
  const float* bb_w = (const float*)d_in[24];
  const float* bb_b = (const float*)d_in[25];
  const float* ff1_w = (const float*)d_in[26];
  const float* ff1_b = (const float*)d_in[27];
  const float* ff2_w = (const float*)d_in[28];
  const float* ff2_b = (const float*)d_in[29];
  const float* ta_w = (const float*)d_in[30];
  const float* ta_b = (const float*)d_in[31];
  const float* tb_w = (const float*)d_in[32];
  const float* tb_b = (const float*)d_in[33];
  const float* gate_w = (const float*)d_in[34];
  const float* gate_b = (const float*)d_in[35];
  const float* head_w1 = (const float*)d_in[36];
  const float* head_b1 = (const float*)d_in[37];
  const float* head_w2 = (const float*)d_in[38];
  const float* head_b2 = (const float*)d_in[39];

  char* base = (char*)d_ws;
  size_t off = 0;
  auto alloc = [&](size_t bytes) -> char* {
    char* p = base + off;
    off += (bytes + 255) & ~(size_t)255;
    return p;
  };
  const size_t PREP1_H = 4ull * 162 * 76 * 76 * 8;
  const size_t ACT1_H = 4ull * 162 * 38 * 38 * 32;
  const size_t ACT2_H = 4ull * 162 * 20 * 20 * 64;
  const size_t ACT3_H = 4ull * 160 * 18 * 18 * 64;
  _Float16* prep1 = (_Float16*)alloc(PREP1_H * 2);
  _Float16* act1 = (_Float16*)alloc(ACT1_H * 2);
  _Float16* act2 = (_Float16*)alloc(ACT2_H * 2);
  _Float16* act3 = (_Float16*)alloc(ACT3_H * 2);
  _Float16* wpack1 = (_Float16*)alloc(2ull * 19 * 512 * 2);
  _Float16* wpack2 = (_Float16*)alloc(4ull * 27 * 512 * 2);
  _Float16* wpack3 = (_Float16*)alloc(4ull * 54 * 512 * 2);
  float* bias1 = (float*)alloc(32 * 4);
  float* bias2 = (float*)alloc(64 * 4);
  float* bias3 = (float*)alloc(64 * 4);
  float* enc = (float*)alloc(4ull * 160 * 64 * 4);
  float* meb = (float*)alloc(4ull * 160 * 16 * 4);
  float* gates = (float*)alloc(4ull * 160 * 64 * 4);
  float* seq = (float*)alloc(4ull * 160 * 64 * 4);
  float* gated = (float*)alloc(4ull * 160 * 64 * 4);

  {
    long n = 4L * 162 * 76 * 76;
    pack_frames<<<(unsigned)((n + 255) / 256), 256, 0, stream>>>(frames, prep1);
  }
  {
    long n16 = (long)(ACT1_H * 2 / 16);
    zero16<<<(unsigned)((n16 + 255) / 256), 256, 0, stream>>>((uint4*)act1, n16);
  }
  {
    long n16 = (long)(ACT2_H * 2 / 16);
    zero16<<<(unsigned)((n16 + 255) / 256), 256, 0, stream>>>((uint4*)act2, n16);
  }
  {
    long n = 2L * 19 * 512;
    pack_w<<<(unsigned)((n + 255) / 256), 256, 0, stream>>>(conv1_w, bn1_g, bn1_v, wpack1, 32, 6, 8, 3, 5, 5, 19);
  }
  {
    long n = 4L * 27 * 512;
    pack_w<<<(unsigned)((n + 255) / 256), 256, 0, stream>>>(conv2_w, bn2_g, bn2_v, wpack2, 64, 32, 32, 3, 3, 3, 27);
  }
  {
    long n = 4L * 54 * 512;
    pack_w<<<(unsigned)((n + 255) / 256), 256, 0, stream>>>(conv3_w, bn3_g, bn3_v, wpack3, 64, 64, 64, 3, 3, 3, 54);
  }
  bias_k<<<1, 32, 0, stream>>>(bn1_g, bn1_b, bn1_m, bn1_v, bias1, 32);
  bias_k<<<1, 64, 0, stream>>>(bn2_g, bn2_b, bn2_m, bn2_v, bias2, 64);
  bias_k<<<1, 64, 0, stream>>>(bn3_g, bn3_b, bn3_m, bn3_v, bias3, 64);

  conv1_k<<<4 * 160 * 36, 64, 0, stream>>>(prep1, wpack1, bias1, act1);
  conv2_k<<<4 * 160 * 18, 128, 0, stream>>>(act1, wpack2, bias2, act2);
  conv3_k<<<4 * 160 * 18, 128, 0, stream>>>(act2, wpack3, bias3, act3);

  spatial_mean<<<640, 64, 0, stream>>>(act3, enc);
  motion_k<<<640, 64, 0, stream>>>(pose, flow, ln_g, ln_b, mlp_w1, mlp_b1,
                                   mlp_w2, mlp_b2, gate_w, gate_b, meb, gates);
  cfc_k<<<1, 256, 0, stream>>>(enc, meb, bb_w, bb_b, ff1_w, ff1_b, ff2_w, ff2_b,
                               ta_w, ta_b, tb_w, tb_b, seq);
  gate_k<<<1, 256, 0, stream>>>(seq, gates, gated);
  head_k<<<640, 32, 0, stream>>>(gated, head_w1, head_b1, head_w2, head_b2,
                                 (float*)d_out);
}
